// CharLSTMBlock_33956011442654
// MI455X (gfx1250) — compile-verified, matrix operand order checked by reading, not run on hardware
//
#include <hip/hip_runtime.h>

// CharLSTM: N=32768 sequences, Tc=25, D=32, H=64, 4H=256.
// Per step, per 16-seq tile: z[16,256] = b + [x|h][16,96] @ [W;U][96,256]
// via 3 K-chunks x 16 col-tiles of v_wmma_f32_16x16x32_f16.
// Column tiles processed as 4 gate-groups (i,f,g,o = tiles j,j+4,j+8,j+12):
// 12 B-frags loaded up front per group, then 12 WMMAs back-to-back, then
// branch-free gate math on the trans units (v_exp_f32 / v_rcp_f32).

typedef __attribute__((ext_vector_type(16))) _Float16 v16h;
typedef __attribute__((ext_vector_type(8)))  _Float16 v8h;
typedef __attribute__((ext_vector_type(4)))  _Float16 v4h;
typedef __attribute__((ext_vector_type(8)))  float    v8f;

#define N_SEQ   32768   // B*TW
#define TC      25
#define DIM     32
#define HID     64
#define ZCOLS   256     // 4H
#define WAVES   4       // waves (16-seq tiles) per block
#define BLOCK   (WAVES * 32)
#define NTILES  (N_SEQ / 16)
#define GRID    (NTILES / WAVES)
#define ZP_ELEMS (3 * 16 * 32 * 16)   // [kc][ct][lane][half] f16 fragments

#define WAITDS() asm volatile("s_wait_dscnt 0x0" ::: "memory")

// Branch-free activations on the transcendental units.
__device__ __forceinline__ float sigmoid_fast(float x) {
    // 1 / (1 + e^-x)
    return __builtin_amdgcn_rcpf(1.0f + __expf(-x));
}
__device__ __forceinline__ float tanh_fast(float x) {
    // tanh(x) = 1 - 2/(1 + e^{2x})
    return 1.0f - 2.0f * __builtin_amdgcn_rcpf(1.0f + __expf(2.0f * x));
}

#define WMMA(A, B, C) __builtin_amdgcn_wmma_f32_16x16x32_f16(false, (A), false, (B), (short)0, (C), false, false)

// ---------------------------------------------------------------------------
// Pack [W;U] (f32, row-major [96][256]) into f16 B-fragments:
// zp[((kc*16+ct)*32+lane)*16 + i] = Z[kc*32 + kbase + koff(i)][ct*16 + (lane&15)]
// kbase = (lane>>4)*8 ; koff(i) = i<8 ? i : 8+i   (doc 16-bit fragment layout)
// ---------------------------------------------------------------------------
__global__ void charlstm_pack_weights(const float* __restrict__ W,
                                      const float* __restrict__ U,
                                      _Float16* __restrict__ zp) {
    int idx = blockIdx.x * blockDim.x + threadIdx.x;
    if (idx >= ZP_ELEMS) return;
    int i    =  idx        & 15;
    int lane = (idx >> 4)  & 31;
    int ct   = (idx >> 9)  & 15;
    int kc   =  idx >> 13;
    int kbase = (lane >> 4) * 8;
    int k = kbase + (i < 8 ? i : 8 + i);         // 0..31 within chunk
    int n = ct * 16 + (lane & 15);
    float v;
    if      (kc == 0) v = W[k * ZCOLS + n];           // x dims 0..31
    else if (kc == 1) v = U[k * ZCOLS + n];           // h dims 0..31
    else              v = U[(32 + k) * ZCOLS + n];    // h dims 32..63
    zp[idx] = (_Float16)v;
}

// ---------------------------------------------------------------------------
// Main fused LSTM kernel. One wave per 16-sequence tile, 4 waves per block.
// ---------------------------------------------------------------------------
__global__ __launch_bounds__(BLOCK, 1)
void charlstm_main(const int*   __restrict__ chars,
                   const float* __restrict__ E,
                   const float* __restrict__ bias,
                   const _Float16* __restrict__ zp,
                   float* __restrict__ out) {
    __shared__ __align__(32) _Float16 zlds[ZP_ELEMS];          // 48 KB weights
    __shared__ __align__(32) _Float16 xbuf[WAVES][16][DIM];    // 4 KB x staging
    __shared__ __align__(32) _Float16 hbuf[WAVES][16][HID];    // 8 KB h staging

    // Stage packed weights block-wide (L2 -> LDS, 3072 x int4).
    {
        const int4* src = (const int4*)zp;
        int4* dst = (int4*)zlds;
        for (int i = threadIdx.x; i < ZP_ELEMS / 8; i += BLOCK) dst[i] = src[i];
    }
    __syncthreads();

    const int lane  = threadIdx.x & 31;
    const int wave  = threadIdx.x >> 5;
    const int n0    = (blockIdx.x * WAVES + wave) * 16;   // first sequence of tile
    const int nl    = lane & 15;
    const int half  = lane >> 4;
    const int kbase = half * 8;   // A-fragment K base for this lane
    const int mbase = half * 8;   // C/D-fragment row base for this lane

    // h staging starts at zero (h0 = 0): 4 x v8h per lane
    {
        v8h z8 = {};
        _Float16* hb = &hbuf[wave][0][0];
        for (int i = lane * 8; i < 16 * HID; i += 32 * 8)
            *(v8h*)&hb[i] = z8;
    }

    // Bias, replicated down the 8 rows of each accumulator tile.
    float bias_r[16];
#pragma unroll
    for (int ct = 0; ct < 16; ++ct) bias_r[ct] = bias[ct * 16 + nl];

    // Cell state: 4 gate-groups x 8 rows per lane, register-resident.
    float cstate[4][8];
#pragma unroll
    for (int j = 0; j < 4; ++j)
#pragma unroll
        for (int e = 0; e < 8; ++e) cstate[j][e] = 0.0f;

    union V16 { v16h v; v8h h[2]; };

    for (int t = 0; t < TC; ++t) {
        // ---- embed: lane (s=nl, half) stages E[chars[n0+s, t]][half*16..+16) ----
        {
            int cid = chars[(n0 + nl) * TC + t];
            const float4* er4 = (const float4*)(E + cid * DIM + half * 16);
#pragma unroll
            for (int q = 0; q < 4; ++q) {
                float4 v = er4[q];
                v4h hx;
                hx[0] = (_Float16)v.x; hx[1] = (_Float16)v.y;
                hx[2] = (_Float16)v.z; hx[3] = (_Float16)v.w;
                *(v4h*)&xbuf[wave][nl][half * 16 + q * 4] = hx;
            }
        }
        WAITDS();   // xbuf writes (and previous step's hbuf writes) visible

        // ---- A fragments: chunk0 = x[0:32], chunk1 = h[0:32], chunk2 = h[32:64]
        V16 a0, a1, a2;
        a0.h[0] = *(const v8h*)&xbuf[wave][nl][kbase];
        a0.h[1] = *(const v8h*)&xbuf[wave][nl][16 + kbase];
        a1.h[0] = *(const v8h*)&hbuf[wave][nl][kbase];
        a1.h[1] = *(const v8h*)&hbuf[wave][nl][16 + kbase];
        a2.h[0] = *(const v8h*)&hbuf[wave][nl][32 + kbase];
        a2.h[1] = *(const v8h*)&hbuf[wave][nl][48 + kbase];
        v16h afrag[3] = { a0.v, a1.v, a2.v };

        // ---- 4 gate groups: tiles (j, j+4, j+8, j+12) = (i, f, g, o) ----
#pragma unroll
        for (int j = 0; j < 4; ++j) {
            v8f zi, zf, zg, zo;
#pragma unroll
            for (int e = 0; e < 8; ++e) {
                zi[e] = bias_r[j];
                zf[e] = bias_r[j + 4];
                zg[e] = bias_r[j + 8];
                zo[e] = bias_r[j + 12];
            }

            // Load ALL 12 B-fragments of this group, then 12 WMMAs back-to-back.
            v16h bfr[12];
#pragma unroll
            for (int kc = 0; kc < 3; ++kc) {
                const _Float16* zb = &zlds[(kc * 16 * 32 + lane) * 16];
                bfr[kc * 4 + 0] = *(const v16h*)&zb[(j     ) * 32 * 16];
                bfr[kc * 4 + 1] = *(const v16h*)&zb[(j +  4) * 32 * 16];
                bfr[kc * 4 + 2] = *(const v16h*)&zb[(j +  8) * 32 * 16];
                bfr[kc * 4 + 3] = *(const v16h*)&zb[(j + 12) * 32 * 16];
            }
#pragma unroll
            for (int kc = 0; kc < 3; ++kc) {
                v16h a = afrag[kc];
                zi = WMMA(a, bfr[kc * 4 + 0], zi);
                zf = WMMA(a, bfr[kc * 4 + 1], zf);
                zg = WMMA(a, bfr[kc * 4 + 2], zg);
                zo = WMMA(a, bfr[kc * 4 + 3], zo);
            }

            // ---- branch-free gates; h written to LDS for next step's A ----
            float hv8[8];
#pragma unroll
            for (int e = 0; e < 8; ++e) {
                float iv = sigmoid_fast(zi[e]);
                float fv = sigmoid_fast(zf[e]);
                float gv = tanh_fast(zg[e]);
                float ov = sigmoid_fast(zo[e]);
                float cc = fv * cstate[j][e] + iv * gv;
                cstate[j][e] = cc;
                float hv = ov * tanh_fast(cc);
                hv8[e] = hv;
                hbuf[wave][mbase + e][j * 16 + nl] = (_Float16)hv;
            }
            if (t == TC - 1) {
#pragma unroll
                for (int e = 0; e < 8; ++e)
                    out[(size_t)(n0 + mbase + e) * HID + j * 16 + nl] = hv8[e];
            }
        }
    }
}

extern "C" void kernel_launch(void* const* d_in, const int* in_sizes, int n_in,
                              void* d_out, int out_size, void* d_ws, size_t ws_size,
                              hipStream_t stream) {
    (void)in_sizes; (void)n_in; (void)out_size; (void)ws_size;
    const int*   chars = (const int*)d_in[0];
    const float* E     = (const float*)d_in[1];
    const float* W     = (const float*)d_in[2];
    const float* U     = (const float*)d_in[3];
    const float* b     = (const float*)d_in[4];
    float* out  = (float*)d_out;
    _Float16* zp = (_Float16*)d_ws;   // 48 KB packed-fragment weights

    charlstm_pack_weights<<<(ZP_ELEMS + 255) / 256, 256, 0, stream>>>(W, U, zp);
    charlstm_main<<<GRID, BLOCK, 0, stream>>>(chars, E, b, zp, out);
}
